// IDUCell_2181843387041
// MI455X (gfx1250) — matrix operand order repaired and checked by reading, not compiled, not run on hardware
//
#include <hip/hip_runtime.h>
#include <hip/hip_bf16.h>
#include <math.h>

typedef __attribute__((ext_vector_type(2)))  float  v2f;
typedef __attribute__((ext_vector_type(8)))  float  v8f;
typedef __attribute__((ext_vector_type(8)))  __bf16 v8bf;
typedef __attribute__((ext_vector_type(16))) __bf16 v16bf;

__device__ __forceinline__ v8f wmma_f32(v2f a, v2f b, v8f c) {
    // (neg_a, A, neg_b, B, c_mod, C, reuse_a, reuse_b)
    return __builtin_amdgcn_wmma_f32_16x16x4_f32(false, a, false, b, (short)0, c, false, false);
}

__device__ __forceinline__ v8f wmma_bf16(v16bf a, v16bf b, v8f c) {
    return __builtin_amdgcn_wmma_f32_16x16x32_bf16(false, a, false, b, (short)0, c, false, false);
}

// Assemble a 16-element bf16 fragment from two aligned 8-element LDS chunks.
__device__ __forceinline__ v16bf load16(const __bf16* base, int o1, int o2) {
    v8bf a = *(const v8bf*)(base + o1);
    v8bf b = *(const v8bf*)(base + o2);
    return __builtin_shufflevector(a, b, 0,1,2,3,4,5,6,7,8,9,10,11,12,13,14,15);
}

__device__ __forceinline__ unsigned pack_bf16(__bf16 a, __bf16 b) {
    return (unsigned)__builtin_bit_cast(unsigned short, a)
         | ((unsigned)__builtin_bit_cast(unsigned short, b) << 16);
}

// Low 32 bits of a flat shared-memory address are the LDS byte offset
// (ISA 10.2).
__device__ __forceinline__ unsigned lds_off(const void* p) {
    return (unsigned)(unsigned long long)p;
}

// CDNA5 async global->LDS copy, 16 bytes per lane, tracked on ASYNCcnt.
__device__ __forceinline__ void async_b128(unsigned lds_byte_off, const float* g) {
    unsigned long long ga = (unsigned long long)g;
    asm volatile("global_load_async_to_lds_b128 %0, %1, off"
                 :: "v"(lds_byte_off), "v"(ga) : "memory");
}
__device__ __forceinline__ void async_wait0() {
    asm volatile("s_wait_asynccnt 0" ::: "memory");
}

// ---------------------------------------------------------------------------
// bf16x3 split-precision GEMM: C = act(A@W + bias), fp32 in/out, ~fp32 accuracy
// (hi+lo bf16 split, rel err ~2^-16). A (M,K) lda; W (K,N) ldw; C (M,N) ldc.
// M%64==0, K%32==0, N masked. Block 256 thr = 8 waves; tile 64x64; K chunk 32.
// Per chunk per wave: 6x v_wmma_f32_16x16x32_bf16.
// ---------------------------------------------------------------------------
template<int ACT>
__global__ __launch_bounds__(256, 2)
void gemm_bias_act_bf3(const float* __restrict__ A, long lda,
                       const float* __restrict__ W, long ldw,
                       const float* __restrict__ bias,
                       float* __restrict__ C, long ldc,
                       int N, int K)
{
    constexpr int LDT = 40;                 // 32 + 8 pad bf16; 80B row keeps 16B align
    __shared__ __bf16 Ahi[64 * LDT];
    __shared__ __bf16 Alo[64 * LDT];
    __shared__ __bf16 Bhi[64 * LDT];        // stored transposed: [n][k]
    __shared__ __bf16 Blo[64 * LDT];

    const int tid  = threadIdx.x;
    const int wave = tid >> 5, lane = tid & 31;
    const int wm   = wave & 3,  wn   = wave >> 2;
    const int half = lane >> 4, l15  = lane & 15;
    const long bm  = (long)blockIdx.y * 64;
    const int  bn  = blockIdx.x * 64;

    v8f acc0 = {}, acc1 = {};

    for (int k0 = 0; k0 < K; k0 += 32) {
        // Stage A 64x32: split fp32 -> bf16 hi/lo, packed u32 LDS stores
        {
            int row = tid >> 2;
            int col = (tid & 3) << 3;
            const float* src = A + (bm + row) * lda + (long)(k0 + col);
            #pragma unroll
            for (int j = 0; j < 8; j += 2) {
                float  x0 = src[j], x1 = src[j + 1];
                __bf16 h0 = (__bf16)x0, h1 = (__bf16)x1;
                __bf16 l0 = (__bf16)(x0 - (float)h0);
                __bf16 l1 = (__bf16)(x1 - (float)h1);
                *(unsigned*)&Ahi[row * LDT + col + j] = pack_bf16(h0, h1);
                *(unsigned*)&Alo[row * LDT + col + j] = pack_bf16(l0, l1);
            }
        }
        // Stage B 32x64 transposed to [n][k]: thread owns a 2k x 4n micro-block
        // so transposed stores pack k-pairs into u32 (even k => 4B aligned).
        {
            int krow = (tid >> 4) << 1;    // 0,2,...,30
            int col  = (tid & 15) << 2;    // 0..60
            const float* s0 = W + (long)(k0 + krow) * ldw + (long)bn + col;
            const float* s1 = s0 + ldw;
            #pragma unroll
            for (int j = 0; j < 4; ++j) {
                bool  ok = (bn + col + j) < N;
                float x0 = ok ? s0[j] : 0.0f;
                float x1 = ok ? s1[j] : 0.0f;
                __bf16 h0 = (__bf16)x0, h1 = (__bf16)x1;
                __bf16 l0 = (__bf16)(x0 - (float)h0);
                __bf16 l1 = (__bf16)(x1 - (float)h1);
                *(unsigned*)&Bhi[(col + j) * LDT + krow] = pack_bf16(h0, h1);
                *(unsigned*)&Blo[(col + j) * LDT + krow] = pack_bf16(l0, l1);
            }
        }
        __syncthreads();

        // A frag (16x32 bf16): lane half selects K {8h..8h+7} and {16+8h..+7}
        const __bf16* arh = &Ahi[(wm * 16 + l15) * LDT];
        const __bf16* arl = &Alo[(wm * 16 + l15) * LDT];
        const int o1 = 8 * half, o2 = 16 + 8 * half;
        v16bf ah = load16(arh, o1, o2);
        v16bf al = load16(arl, o1, o2);

        // B frag (32x16 bf16): lane n = l15, K {8h..8h+7} and {16+8h..+7} (transposed LDS)
        const int bo = 8 * half;
        v16bf b0h = load16(&Bhi[(wn * 32 + l15) * LDT], bo, bo + 16);
        v16bf b0l = load16(&Blo[(wn * 32 + l15) * LDT], bo, bo + 16);
        v16bf b1h = load16(&Bhi[(wn * 32 + 16 + l15) * LDT], bo, bo + 16);
        v16bf b1l = load16(&Blo[(wn * 32 + 16 + l15) * LDT], bo, bo + 16);

        acc0 = wmma_bf16(ah, b0h, acc0);
        acc0 = wmma_bf16(ah, b0l, acc0);
        acc0 = wmma_bf16(al, b0h, acc0);
        acc1 = wmma_bf16(ah, b1h, acc1);
        acc1 = wmma_bf16(ah, b1l, acc1);
        acc1 = wmma_bf16(al, b1h, acc1);
        __syncthreads();
    }

    #pragma unroll
    for (int v = 0; v < 8; ++v) {
        long m  = bm + wm * 16 + v + 8 * half;
        int  n0 = bn + wn * 32 + l15;
        int  n1 = n0 + 16;
        if (n0 < N) {
            float r = acc0[v] + bias[n0];
            if (ACT == 1) r = fmaxf(r, 0.0f);
            C[m * ldc + n0] = r;
        }
        if (n1 < N) {
            float r = acc1[v] + bias[n1];
            if (ACT == 1) r = fmaxf(r, 0.0f);
            C[m * ldc + n1] = r;
        }
    }
}

// ---------------------------------------------------------------------------
// Exact-fp32 tiled GEMM (V_WMMA_F32_16X16X4_F32) for small/precision paths.
// ---------------------------------------------------------------------------
template<int ACT>
__global__ __launch_bounds__(256, 2)
void gemm_bias_act(const float* __restrict__ A, long lda,
                   const float* __restrict__ W, long ldw,
                   const float* __restrict__ bias,
                   float* __restrict__ C, long ldc,
                   int N, int K)
{
    __shared__ float As[64][17];
    __shared__ float Bs[16][65];

    const int tid  = threadIdx.x;
    const int wave = tid >> 5, lane = tid & 31;
    const int wm   = wave & 3,  wn   = wave >> 2;
    const int half = lane >> 4, l15  = lane & 15;
    const long bm  = (long)blockIdx.y * 64;
    const int  bn  = blockIdx.x * 64;

    v8f acc0 = {}, acc1 = {};

    for (int k0 = 0; k0 < K; k0 += 16) {
        {
            int row = tid >> 2;
            int col = (tid & 3) << 2;
            const float4 v = *reinterpret_cast<const float4*>(
                A + (bm + row) * lda + (long)(k0 + col));
            As[row][col + 0] = v.x;
            As[row][col + 1] = v.y;
            As[row][col + 2] = v.z;
            As[row][col + 3] = v.w;
        }
        {
            int row = tid >> 4;
            int col = (tid & 15) << 2;
            const float* src = W + (long)(k0 + row) * ldw + (long)bn + col;
            #pragma unroll
            for (int j = 0; j < 4; ++j) {
                int n = bn + col + j;
                Bs[row][col + j] = (n < N) ? src[j] : 0.0f;
            }
        }
        __syncthreads();

        #pragma unroll
        for (int kk = 0; kk < 4; ++kk) {
            const int kb = kk * 4 + 2 * half;
            v2f a, b0, b1;
            a.x  = As[wm * 16 + l15][kb];
            a.y  = As[wm * 16 + l15][kb + 1];
            b0.x = Bs[kb][wn * 32 + l15];
            b0.y = Bs[kb + 1][wn * 32 + l15];
            b1.x = Bs[kb][wn * 32 + 16 + l15];
            b1.y = Bs[kb + 1][wn * 32 + 16 + l15];
            acc0 = wmma_f32(a, b0, acc0);
            acc1 = wmma_f32(a, b1, acc1);
        }
        __syncthreads();
    }

    #pragma unroll
    for (int v = 0; v < 8; ++v) {
        long m  = bm + wm * 16 + v + 8 * half;
        int  n0 = bn + wn * 32 + l15;
        int  n1 = n0 + 16;
        if (n0 < N) {
            float r = acc0[v] + bias[n0];
            if (ACT == 1) r = fmaxf(r, 0.0f);
            C[m * ldc + n0] = r;
        }
        if (n1 < N) {
            float r = acc1[v] + bias[n1];
            if (ACT == 1) r = fmaxf(r, 0.0f);
            C[m * ldc + n1] = r;
        }
    }
}

// ---------------------------------------------------------------------------
// One fused recurrence step (exact fp32 WMMA), async-to-LDS staged:
//   rt = sigmoid(h@Whr + bhr + x0r); zt = sigmoid(xtz[s] + x0z)
//   ht1 = tanh(xth1[s] + rt*(h@Wh1h1 + bh1h1)); ht = (1-zt)*ht1 + zt*h
// Block 128 thr = 4 waves; block = 16 cols x all 64 rows; K chunk 32.
// h tile + both weight tiles staged once per block via ASYNCcnt DMA
// (removes the 4x per-wave duplication of column-strided weight loads).
// ---------------------------------------------------------------------------
__global__ __launch_bounds__(128, 4)
void gru_step(const float* __restrict__ hprev, long ldh,
              const float* __restrict__ Whr,   const float* __restrict__ bhr,
              const float* __restrict__ Wh1h1, const float* __restrict__ bh1h1,
              const float* __restrict__ x0r,   const float* __restrict__ x0z,
              const float* __restrict__ xtz_all, const float* __restrict__ xth1_all,
              float* __restrict__ hts, int s)
{
    const int H = 1024, S = 64;
    constexpr int LA = 36;                  // 32 + 4 pad floats (144B rows, 16B aligned)
    constexpr int LB = 20;                  // 16 + 4 pad floats (80B rows, 16B aligned)
    __shared__ float As[64 * LA];
    __shared__ float Br[32 * LB];
    __shared__ float Bh[32 * LB];

    const int tid   = threadIdx.x;
    const int mtile = tid >> 5;
    const int lane  = tid & 31;
    const int half  = lane >> 4;
    const int l15   = lane & 15;
    const int n0    = blockIdx.x * 16;

    v8f accR = {}, accH = {};

    for (int k0 = 0; k0 < H; k0 += 32) {
        // async stage h chunk (64 rows x 32 k): 4x b128 per thread
        {
            int r  = tid >> 1;
            int cb = (tid & 1) << 4;
            const float* g = hprev + (long)r * ldh + k0 + cb;
            #pragma unroll
            for (int i = 0; i < 4; ++i)
                async_b128(lds_off(&As[r * LA + cb + i * 4]), g + i * 4);
        }
        // async stage Whr / Wh1h1 chunks (32 k x 16 n each): 1x b128 per thread each
        {
            int k = tid >> 2;
            int c = (tid & 3) << 2;
            async_b128(lds_off(&Br[k * LB + c]), Whr   + (long)(k0 + k) * H + n0 + c);
            async_b128(lds_off(&Bh[k * LB + c]), Wh1h1 + (long)(k0 + k) * H + n0 + c);
        }
        async_wait0();
        __syncthreads();

        #pragma unroll
        for (int kk = 0; kk < 8; ++kk) {
            const int kb = kk * 4 + 2 * half;
            v2f a, br, bh;
            a.x  = As[(mtile * 16 + l15) * LA + kb];
            a.y  = As[(mtile * 16 + l15) * LA + kb + 1];
            br.x = Br[kb * LB + l15];
            br.y = Br[(kb + 1) * LB + l15];
            bh.x = Bh[kb * LB + l15];
            bh.y = Bh[(kb + 1) * LB + l15];
            accR = wmma_f32(a, br, accR);
            accH = wmma_f32(a, bh, accH);
        }
        __syncthreads();
    }

    #pragma unroll
    for (int v = 0; v < 8; ++v) {
        int  b   = mtile * 16 + v + 8 * half;
        int  n   = n0 + l15;
        long bn  = (long)b * H + n;
        long bsn = ((long)b * S + s) * H + n;
        float rt  = 1.0f / (1.0f + __expf(-(accR[v] + bhr[n] + x0r[bn])));
        float zt  = 1.0f / (1.0f + __expf(-(xtz_all[bsn] + x0z[bn])));
        float h1  = accH[v] + bh1h1[n];
        float ht1 = tanhf(xth1_all[bsn] + rt * h1);
        float hp  = hprev[(long)b * ldh + n];
        hts[bsn] = (1.0f - zt) * ht1 + zt * hp;
    }
}

// ---------------------------------------------------------------------------
// Broadcast epilogue: p0es[b,s,:] = ptes[b,S-1,:]; x0es[b,s,:] = xtes[b,S-1,:]
// ---------------------------------------------------------------------------
__global__ void bcast_kernel(const float* __restrict__ xtes,
                             const float* __restrict__ ptes,
                             float* __restrict__ x0es,
                             float* __restrict__ p0es)
{
    const int  S = 64, H = 1024, C = 31;
    const long NH = 64L * 64 * 1024;
    const long NC = 64L * 64 * 31;
    long i = (long)blockIdx.x * blockDim.x + threadIdx.x;
    if (i < NH) {
        long b = i / ((long)S * H);
        long h = i % H;
        x0es[i] = xtes[(b * S + (S - 1)) * (long)H + h];
    } else if (i < NH + NC) {
        long j = i - NH;
        long b = j / ((long)S * C);
        long c = j % C;
        p0es[j] = ptes[(b * S + (S - 1)) * (long)C + c];
    }
}

extern "C" void kernel_launch(void* const* d_in, const int* in_sizes, int n_in,
                              void* d_out, int out_size, void* d_ws, size_t ws_size,
                              hipStream_t stream) {
    const float* x     = (const float*)d_in[0];
    const float* h0    = (const float*)d_in[1];
    const float* Wxe   = (const float*)d_in[2];
    const float* bxe   = (const float*)d_in[3];
    const float* Wep   = (const float*)d_in[4];
    const float* bep   = (const float*)d_in[5];
    const float* Whr   = (const float*)d_in[6];
    const float* bhr   = (const float*)d_in[7];
    const float* Wx0r  = (const float*)d_in[8];
    const float* bx0r  = (const float*)d_in[9];
    const float* Wxtz  = (const float*)d_in[10];
    const float* bxtz  = (const float*)d_in[11];
    const float* Wx0z  = (const float*)d_in[12];
    const float* bx0z  = (const float*)d_in[13];
    const float* Wxth1 = (const float*)d_in[14];
    const float* bxth1 = (const float*)d_in[15];
    const float* Wh1h1 = (const float*)d_in[16];
    const float* bh1h1 = (const float*)d_in[17];

    const int  Bn = 64, S = 64, F = 4096, H = 1024, Cn = 31;
    const long BSH = (long)Bn * S * H;     // 4,194,304
    const long BSC = (long)Bn * S * Cn;    //   126,976

    // d_out layout (return order): hts | ptes | p0es | xtes | x0es
    float* out  = (float*)d_out;
    float* hts  = out;
    float* ptes = out + BSH;
    float* p0es = out + BSH + BSC;
    float* xtes = out + BSH + 2 * BSC;
    float* x0es = out + 2 * BSH + 2 * BSC;

    // scratch: xtz_all | xth1_all | x0r | x0z  (~32.5 MB)
    float* ws       = (float*)d_ws;
    float* xtz_all  = ws;
    float* xth1_all = ws + BSH;
    float* x0r      = ws + 2 * BSH;
    float* x0z      = x0r + (long)Bn * H;

    const float* x0e = xtes + (long)(S - 1) * H;  // strided view, lda = S*H
    const long   ld0 = (long)S * H;

    // 1) xtes = relu(X @ Wxe + bxe)       M=4096 K=4096 N=1024  (bf16x3)
    gemm_bias_act_bf3<1><<<dim3(H / 64, (Bn * S) / 64), 256, 0, stream>>>(
        x, (long)F, Wxe, (long)H, bxe, xtes, (long)H, H, F);
    // 2) ptes = xtes @ Wep + bep          N=31 masked           (fp32 WMMA)
    gemm_bias_act<0><<<dim3(1, (Bn * S) / 64), 256, 0, stream>>>(
        xtes, (long)H, Wep, (long)Cn, bep, ptes, (long)Cn, Cn, H);
    // 3) xtz_all = xtes @ Wxtz + bxtz                           (bf16x3)
    gemm_bias_act_bf3<0><<<dim3(H / 64, (Bn * S) / 64), 256, 0, stream>>>(
        xtes, (long)H, Wxtz, (long)H, bxtz, xtz_all, (long)H, H, H);
    // 4) xth1_all = xtes @ Wxth1 + bxth1                        (bf16x3)
    gemm_bias_act_bf3<0><<<dim3(H / 64, (Bn * S) / 64), 256, 0, stream>>>(
        xtes, (long)H, Wxth1, (long)H, bxth1, xth1_all, (long)H, H, H);
    // 5) x0r = x0e @ Wx0r + bx0r          M=64                  (fp32 WMMA)
    gemm_bias_act<0><<<dim3(H / 64, 1), 256, 0, stream>>>(
        x0e, ld0, Wx0r, (long)H, bx0r, x0r, (long)H, H, H);
    // 6) x0z = x0e @ Wx0z + bx0z                                (fp32 WMMA)
    gemm_bias_act<0><<<dim3(H / 64, 1), 256, 0, stream>>>(
        x0e, ld0, Wx0z, (long)H, bx0z, x0z, (long)H, H, H);
    // 7) broadcast p0es / x0es from the s = S-1 slices
    {
        long total  = BSH + BSC;
        int  blocks = (int)((total + 255) / 256);
        bcast_kernel<<<dim3(blocks), 256, 0, stream>>>(xtes, ptes, x0es, p0es);
    }
    // 8) sequential recurrence: 64 fused-step launches (graph-capturable)
    for (int s = 0; s < S; ++s) {
        const float* hp  = (s == 0) ? h0 : (hts + (long)(s - 1) * H);
        const long   ldh = (s == 0) ? (long)H : (long)S * H;
        gru_step<<<dim3(H / 16), 128, 0, stream>>>(
            hp, ldh, Whr, bhr, Wh1h1, bh1h1, x0r, x0z, xtz_all, xth1_all, hts, s);
    }
}
